// MoModel_25658134626617
// MI455X (gfx1250) — compile-verified, matrix operand order checked by reading, not run on hardware
//
#include <hip/hip_runtime.h>
#include <hip/hip_bf16.h>

// ---------------------------------------------------------------------------
// CDNA5 (gfx1250) implementation. wave32. GEMMs via v_wmma_f32_16x16x32_bf16
// (split-bf16 hi+lo for accuracy-sensitive paths). Adjacency kernel stages its
// tiles with the Tensor Data Mover (tensor_load_to_lds + s_wait_tensorcnt)
// using the 6-arg builtin form, with TDM-injected LDS padding.
// ---------------------------------------------------------------------------

typedef __attribute__((ext_vector_type(16))) __bf16 v16bf;
typedef __attribute__((ext_vector_type(8)))  float  v8f;
typedef __attribute__((ext_vector_type(4)))  unsigned int v4u;
typedef __attribute__((ext_vector_type(8)))  int v8i;
typedef __attribute__((ext_vector_type(4)))  int v4i;

#define LDK 40  // padded LDS row stride (in elements) for a 32-wide K chunk

__device__ __forceinline__ v8f wmma_bf16(v16bf a, v16bf b, v8f c) {
  // (neg_a, A, neg_b, B, c_mod, C, reuse_a, reuse_b)
  return __builtin_amdgcn_wmma_f32_16x16x32_bf16(false, a, false, b, (short)0, c, false, false);
}

// Fragment load for 16x16x32 bf16 WMMA, A-operand layout (ISA 7.12.2).
// Row-major operand staged in LDS as [row][k] halfwords with stride LDK.
// lane L: row = base + (L&15); k-runs: [h8 .. h8+7] and [16+h8 .. +7], h8=(L>>4)*8.
__device__ __forceinline__ v16bf frag_ld(const __bf16* s, int rowBase, int lane) {
  int r  = rowBase + (lane & 15);
  int h8 = (lane >> 4) << 3;
  union { v16bf v; uint4 q[2]; } f;
  const __bf16* p = s + r * LDK;
  f.q[0] = *(const uint4*)(p + h8);
  f.q[1] = *(const uint4*)(p + 16 + h8);
  return f.v;
}

// Same fragment pattern but from an f32 LDS tile (stride LDK floats), with
// register-side f32 -> bf16 conversion (used by the TDM-staged adjacency path).
__device__ __forceinline__ v16bf frag_ld_f32(const float* s, int rowBase, int lane) {
  int r  = rowBase + (lane & 15);
  int h8 = (lane >> 4) << 3;
  const float* p = s + r * LDK;
  float4 a0 = *(const float4*)(p + h8);
  float4 a1 = *(const float4*)(p + h8 + 4);
  float4 b0 = *(const float4*)(p + 16 + h8);
  float4 b1 = *(const float4*)(p + 16 + h8 + 4);
  union { v16bf v; __bf16 e[16]; } f;
  f.e[0]  = (__bf16)a0.x; f.e[1]  = (__bf16)a0.y; f.e[2]  = (__bf16)a0.z; f.e[3]  = (__bf16)a0.w;
  f.e[4]  = (__bf16)a1.x; f.e[5]  = (__bf16)a1.y; f.e[6]  = (__bf16)a1.z; f.e[7]  = (__bf16)a1.w;
  f.e[8]  = (__bf16)b0.x; f.e[9]  = (__bf16)b0.y; f.e[10] = (__bf16)b0.z; f.e[11] = (__bf16)b0.w;
  f.e[12] = (__bf16)b1.x; f.e[13] = (__bf16)b1.y; f.e[14] = (__bf16)b1.z; f.e[15] = (__bf16)b1.w;
  return f.v;
}

__device__ __forceinline__ void split_bf16(float x, __bf16& hi, __bf16& lo) {
  hi = (__bf16)x;
  lo = (__bf16)(x - (float)hi);
}

__device__ __forceinline__ float sigmoidf(float x) { return 1.0f / (1.0f + __expf(-x)); }

__device__ __forceinline__ unsigned fkey(float f) {  // monotonic float -> uint
  unsigned u = __float_as_uint(f);
  return (u & 0x80000000u) ? ~u : (u | 0x80000000u);
}

// ---------------------------------------------------------------------------
// Generic split-bf16 WMMA GEMM: C[m,n] = epi( sum_k A[m,k]*Bt[n,k] (+ bias[n]) )
// Block: 256 thr (8 waves, 4x2), tile 128(M) x 64(N), K step 32.
// EPI: 0 = store, 1 = +bias, 2 = sigmoid(+bias)
// ---------------------------------------------------------------------------
template <int EPI>
__global__ __launch_bounds__(256) void gemm_bf16split(
    const float* __restrict__ A, const float* __restrict__ Bt,
    const float* __restrict__ bias, float* __restrict__ C,
    int Kd, int lda, int ldb, int ldc) {
  __shared__ __bf16 Ah[128 * LDK], Al[128 * LDK], Bh[64 * LDK], Bl[64 * LDK];
  const int tid = threadIdx.x, lane = tid & 31, wid = tid >> 5;
  const int wm = wid & 3, wn = wid >> 2;
  const int m0 = blockIdx.y * 128, n0 = blockIdx.x * 64;

  v8f vz = {};
  v8f acc[2][2];
  acc[0][0] = vz; acc[0][1] = vz; acc[1][0] = vz; acc[1][1] = vz;

  for (int k0 = 0; k0 < Kd; k0 += 32) {
    // stage A 128x32 (f32 -> hi/lo bf16)
    for (int it = 0; it < 4; ++it) {
      int slot = tid + it * 256, row = slot >> 3, c4 = (slot & 7) << 2;
      const float* gp = A + (size_t)(m0 + row) * lda + k0 + c4;
      if (k0 + 32 < Kd) __builtin_prefetch(gp + 32, 0, 1);  // global_prefetch_b8
      float4 v = *(const float4*)gp;
      __bf16 h0, l0, h1, l1, h2, l2, h3, l3;
      split_bf16(v.x, h0, l0); split_bf16(v.y, h1, l1);
      split_bf16(v.z, h2, l2); split_bf16(v.w, h3, l3);
      int o = row * LDK + c4;
      Ah[o] = h0; Ah[o + 1] = h1; Ah[o + 2] = h2; Ah[o + 3] = h3;
      Al[o] = l0; Al[o + 1] = l1; Al[o + 2] = l2; Al[o + 3] = l3;
    }
    // stage B 64x32
    for (int it = 0; it < 2; ++it) {
      int slot = tid + it * 256, row = slot >> 3, c4 = (slot & 7) << 2;
      const float* gp = Bt + (size_t)(n0 + row) * ldb + k0 + c4;
      float4 v = *(const float4*)gp;
      __bf16 h0, l0, h1, l1, h2, l2, h3, l3;
      split_bf16(v.x, h0, l0); split_bf16(v.y, h1, l1);
      split_bf16(v.z, h2, l2); split_bf16(v.w, h3, l3);
      int o = row * LDK + c4;
      Bh[o] = h0; Bh[o + 1] = h1; Bh[o + 2] = h2; Bh[o + 3] = h3;
      Bl[o] = l0; Bl[o + 1] = l1; Bl[o + 2] = l2; Bl[o + 3] = l3;
    }
    __syncthreads();
    v16bf ah[2], al[2], bh[2], bl[2];
    ah[0] = frag_ld(Ah, wm * 32, lane);      ah[1] = frag_ld(Ah, wm * 32 + 16, lane);
    al[0] = frag_ld(Al, wm * 32, lane);      al[1] = frag_ld(Al, wm * 32 + 16, lane);
    bh[0] = frag_ld(Bh, wn * 32, lane);      bh[1] = frag_ld(Bh, wn * 32 + 16, lane);
    bl[0] = frag_ld(Bl, wn * 32, lane);      bl[1] = frag_ld(Bl, wn * 32 + 16, lane);
    for (int i = 0; i < 2; ++i)
      for (int j = 0; j < 2; ++j) {
        acc[i][j] = wmma_bf16(ah[i], bh[j], acc[i][j]);
        acc[i][j] = wmma_bf16(ah[i], bl[j], acc[i][j]);
        acc[i][j] = wmma_bf16(al[i], bh[j], acc[i][j]);
      }
    __syncthreads();
  }

  const int h8 = (lane >> 4) << 3, nl16 = lane & 15;
  for (int i = 0; i < 2; ++i)
    for (int j = 0; j < 2; ++j) {
      int n = n0 + wn * 32 + j * 16 + nl16;
      float bv = (EPI >= 1) ? bias[n] : 0.0f;
      for (int r = 0; r < 8; ++r) {
        int m = m0 + wm * 32 + i * 16 + h8 + r;
        float v = acc[i][j][r] + bv;
        if (EPI == 2) v = sigmoidf(v);
        C[(size_t)m * ldc + n] = v;
      }
    }
}

// ---------------------------------------------------------------------------
// Fused distance GEMM + row argmin: idx[m] = argmin_n (cnorm2[n] - 2 * x_m . cb_n)
// X:[*,512], CB:[1024,512]. One block per 128-row band; loops all 16 N tiles.
// ---------------------------------------------------------------------------
__global__ __launch_bounds__(256) void argmin_gemm(
    const float* __restrict__ X, const float* __restrict__ CB,
    const float* __restrict__ cnorm2, int* __restrict__ idxOut) {
  __shared__ __bf16 Ah[128 * LDK], Al[128 * LDK], Bh[64 * LDK], Bl[64 * LDK];
  __shared__ unsigned long long best[128];
  const int tid = threadIdx.x, lane = tid & 31, wid = tid >> 5;
  const int wm = wid & 3, wn = wid >> 2;
  const int m0 = blockIdx.x * 128;
  if (tid < 128) best[tid] = 0xFFFFFFFFFFFFFFFFULL;
  __syncthreads();

  v8f vz = {};
  for (int nt = 0; nt < 16; ++nt) {
    const int n0 = nt * 64;
    v8f acc[2][2];
    acc[0][0] = vz; acc[0][1] = vz; acc[1][0] = vz; acc[1][1] = vz;
    for (int k0 = 0; k0 < 512; k0 += 32) {
      for (int it = 0; it < 4; ++it) {
        int slot = tid + it * 256, row = slot >> 3, c4 = (slot & 7) << 2;
        const float* gp = X + (size_t)(m0 + row) * 512 + k0 + c4;
        if (k0 < 480) __builtin_prefetch(gp + 32, 0, 1);
        float4 v = *(const float4*)gp;
        __bf16 h0, l0, h1, l1, h2, l2, h3, l3;
        split_bf16(v.x, h0, l0); split_bf16(v.y, h1, l1);
        split_bf16(v.z, h2, l2); split_bf16(v.w, h3, l3);
        int o = row * LDK + c4;
        Ah[o] = h0; Ah[o + 1] = h1; Ah[o + 2] = h2; Ah[o + 3] = h3;
        Al[o] = l0; Al[o + 1] = l1; Al[o + 2] = l2; Al[o + 3] = l3;
      }
      for (int it = 0; it < 2; ++it) {
        int slot = tid + it * 256, row = slot >> 3, c4 = (slot & 7) << 2;
        const float* gp = CB + (size_t)(n0 + row) * 512 + k0 + c4;
        float4 v = *(const float4*)gp;
        __bf16 h0, l0, h1, l1, h2, l2, h3, l3;
        split_bf16(v.x, h0, l0); split_bf16(v.y, h1, l1);
        split_bf16(v.z, h2, l2); split_bf16(v.w, h3, l3);
        int o = row * LDK + c4;
        Bh[o] = h0; Bh[o + 1] = h1; Bh[o + 2] = h2; Bh[o + 3] = h3;
        Bl[o] = l0; Bl[o + 1] = l1; Bl[o + 2] = l2; Bl[o + 3] = l3;
      }
      __syncthreads();
      v16bf ah[2], al[2], bh[2], bl[2];
      ah[0] = frag_ld(Ah, wm * 32, lane);      ah[1] = frag_ld(Ah, wm * 32 + 16, lane);
      al[0] = frag_ld(Al, wm * 32, lane);      al[1] = frag_ld(Al, wm * 32 + 16, lane);
      bh[0] = frag_ld(Bh, wn * 32, lane);      bh[1] = frag_ld(Bh, wn * 32 + 16, lane);
      bl[0] = frag_ld(Bl, wn * 32, lane);      bl[1] = frag_ld(Bl, wn * 32 + 16, lane);
      for (int i = 0; i < 2; ++i)
        for (int j = 0; j < 2; ++j) {
          acc[i][j] = wmma_bf16(ah[i], bh[j], acc[i][j]);
          acc[i][j] = wmma_bf16(ah[i], bl[j], acc[i][j]);
          acc[i][j] = wmma_bf16(al[i], bh[j], acc[i][j]);
        }
      __syncthreads();
    }
    // epilogue: candidate (dist, n) -> packed u64 min into LDS best[] (ds_min_u64)
    const int h8 = (lane >> 4) << 3, nl16 = lane & 15;
    int nA = n0 + wn * 32 + nl16;
    int nB = nA + 16;
    float cA = cnorm2[nA], cB = cnorm2[nB];
    for (int i = 0; i < 2; ++i)
      for (int r = 0; r < 8; ++r) {
        int ml = wm * 32 + i * 16 + h8 + r;
        float dA = cA - 2.0f * acc[i][0][r];
        float dB = cB - 2.0f * acc[i][1][r];
        float d = (dA <= dB) ? dA : dB;
        unsigned n = (unsigned)((dA <= dB) ? nA : nB);
        unsigned long long packed = (((unsigned long long)fkey(d)) << 32) | n;
        atomicMin(&best[ml], packed);
      }
  }
  __syncthreads();
  if (tid < 128) idxOut[m0 + tid] = (int)(best[tid] & 0xFFFFFFFFu);
}

// ---------------------------------------------------------------------------
// Batched adjacency: one block per graph g; out = sigmoid( dot(z_m,z_n) *
// inv[m]*inv[n] / 0.1 ). 128x128 tile, K=512. Tile staging via Tensor Data
// Mover (tensor_load_to_lds, TENSORcnt). Normalization applied in the f32
// epilogue (more accurate than scaling the bf16 operands).
// ---------------------------------------------------------------------------
__global__ __launch_bounds__(256) void adj_kernel(
    const float* __restrict__ Z, const float* __restrict__ inv, float* __restrict__ Aout) {
  __shared__ float Xf[128 * LDK];   // f32 tile, padded row stride 40 floats
  __shared__ float Sinv[128];
  const int tid = threadIdx.x, lane = tid & 31, wid = tid >> 5;
  const int wm = wid >> 1, wn = wid & 1;  // 4 x 2 waves, each 32(M) x 64(N)
  const int g = blockIdx.x;
  const size_t rbase = (size_t)g * 128;

  if (tid < 128) Sinv[tid] = inv[rbase + tid];
  __syncthreads();

  v8f vz = {};
  v8f acc[2][4];
  for (int i = 0; i < 2; ++i) for (int j = 0; j < 4; ++j) acc[i][j] = vz;

  for (int k0 = 0; k0 < 512; k0 += 32) {
    // --- TDM: one wave issues a 2D 128x32-f32 tile DMA into LDS. ---
    if (wid == 0) {
      unsigned long long ga =
          (unsigned long long)(uintptr_t)(Z + rbase * 512 + (size_t)k0);
      unsigned ldsAddr = (unsigned)(uintptr_t)(&Xf[0]);  // low 32 bits = LDS offset
      // D# group0: count=1 | lds_addr | global_addr[56:0] | type=2
      v4u g0;
      g0[0] = 1u;                       // count = 1 valid descriptor
      g0[1] = ldsAddr;                  // lds_addr (bytes)
      g0[2] = (unsigned)ga;             // global_addr[31:0]
      g0[3] = (unsigned)((ga >> 32) & 0x1FFFFFFu) | (2u << 30);  // addr[56:32] | type=2
      // D# group1: data_size=4B, pad_enable, pad_interval=32dw, pad_amount=8dw,
      // tensor_dim0=512, tensor_dim1=65536, tile_dim0=32, tile_dim1=128,
      // tensor_dim0_stride=512.
      v8i g1;
      g1[0] = (int)((2u << 16) | (1u << 20) | (4u << 22) | (7u << 25));
      g1[1] = (int)(512u << 16);        // tensor_dim0[15:0] in [31:16]
      g1[2] = 0;                        // tensor_dim0[31:16]=0, tensor_dim1[15:0]=0
      g1[3] = (int)(1u | (32u << 16));  // tensor_dim1[31:16]=1 (65536), tile_dim0=32
      g1[4] = 128;                      // tile_dim1 = 128, tile_dim2 = 0
      g1[5] = 512;                      // tensor_dim0_stride[31:0]
      g1[6] = 0;
      g1[7] = 0;
      v4i gz4 = {0, 0, 0, 0};                       // groups 2/3: 2-D tensor, unused
      v8i gz8 = {0, 0, 0, 0, 0, 0, 0, 0};           // extra group (6-arg form), unused
      __builtin_amdgcn_tensor_load_to_lds(g0, g1, gz4, gz4, gz8, 0);
      __builtin_amdgcn_s_wait_tensorcnt(0);         // s_wait_tensorcnt 0
    }
    __syncthreads();
    v16bf a[2], b[4];
    a[0] = frag_ld_f32(Xf, wm * 32, lane);      a[1] = frag_ld_f32(Xf, wm * 32 + 16, lane);
    b[0] = frag_ld_f32(Xf, wn * 64, lane);      b[1] = frag_ld_f32(Xf, wn * 64 + 16, lane);
    b[2] = frag_ld_f32(Xf, wn * 64 + 32, lane); b[3] = frag_ld_f32(Xf, wn * 64 + 48, lane);
    for (int i = 0; i < 2; ++i)
      for (int j = 0; j < 4; ++j)
        acc[i][j] = wmma_bf16(a[i], b[j], acc[i][j]);
    __syncthreads();
  }
  const int h8 = (lane >> 4) << 3, nl16 = lane & 15;
  float* out = Aout + (size_t)g * 16384;
  for (int i = 0; i < 2; ++i)
    for (int j = 0; j < 4; ++j) {
      int n = wn * 64 + j * 16 + nl16;
      float sn = Sinv[n] * 10.0f;  // fold 1/TEMP
      for (int r = 0; r < 8; ++r) {
        int m = wm * 32 + i * 16 + h8 + r;
        out[m * 128 + n] = sigmoidf(acc[i][j][r] * Sinv[m] * sn);
      }
    }
}

// ---------------------------------------------------------------------------
// Elementwise & small kernels
// ---------------------------------------------------------------------------
__global__ void prep_causal(const float* __restrict__ causal, float* __restrict__ outDiag,
                            float* __restrict__ outCM, float* __restrict__ AE,
                            float* __restrict__ diagvals) {
  int i = blockIdx.x * blockDim.x + threadIdx.x;  // 1024*1024
  int r = i >> 10, c = i & 1023;
  float v = causal[i];
  bool d = (r == c);
  outDiag[i] = d ? v : 0.0f;
  outCM[i]   = d ? 0.0f : v;
  AE[i]      = d ? 1.0f : v;
  if (d) diagvals[r] = v;
}

__global__ void prep_cs(const float* __restrict__ ct, const float* __restrict__ diagvals,
                        float* __restrict__ cs, float* __restrict__ csT,
                        float* __restrict__ S1T) {
  int i = blockIdx.x * blockDim.x + threadIdx.x;  // 1024*512
  int k = i >> 9, d = i & 511;
  float s = sigmoidf(ct[i]);
  cs[i] = s;
  csT[d * 1024 + k] = s;
  S1T[d * 1024 + k] = diagvals[k] * s;
}

// MODE 0: out[row] = sum(x^2) ; MODE 1: out[row] = 1/max(||x||, 1e-12)
template <int MODE>
__global__ void rowreduce(const float* __restrict__ src, float* __restrict__ out, int cols) {
  int row = blockIdx.x * 8 + (threadIdx.x >> 5);
  int lane = threadIdx.x & 31;
  const float* p = src + (size_t)row * cols;
  float s = 0.0f;
  for (int c = lane; c < cols; c += 32) { float v = p[c]; s += v * v; }
  for (int o = 16; o > 0; o >>= 1) s += __shfl_down(s, o);
  if (lane == 0) out[row] = (MODE == 0) ? s : 1.0f / fmaxf(sqrtf(s), 1e-12f);
}

// One block per graph: gather rows by idx, copy x, write z_nodes, pooled means.
__global__ __launch_bounds__(256) void pool_scatter(
    const float* __restrict__ x, const int* __restrict__ idx,
    const float* __restrict__ ct, const float* __restrict__ ccb,
    const float* __restrict__ cntcb, float* __restrict__ zOut,
    float* __restrict__ xOut, float* __restrict__ pooledCausal,
    float* __restrict__ pooledX, float* __restrict__ pooledCounter) {
  int g = blockIdx.x, t = threadIdx.x;
  int d0 = t, d1 = t + 256;
  float ax0 = 0, ax1 = 0, ac0 = 0, ac1 = 0, an0 = 0, an1 = 0;
  for (int i = 0; i < 128; ++i) {
    int n = g * 128 + i;
    int ii = idx[n];
    size_t xb = (size_t)n * 512, cb = (size_t)ii * 512;
    float x0 = x[xb + d0], x1 = x[xb + d1];
    xOut[xb + d0] = x0; xOut[xb + d1] = x1;
    float z0 = ct[cb + d0], z1 = ct[cb + d1];
    zOut[xb + d0] = z0; zOut[xb + d1] = z1;
    float c0 = ccb[cb + d0], c1 = ccb[cb + d1];
    float q0 = cntcb[cb + d0], q1 = cntcb[cb + d1];
    ax0 += x0; ax1 += x1;
    ac0 += x0 + c0; ac1 += x1 + c1;
    an0 += q0; an1 += q1;
  }
  const float sc = 1.0f / 128.0f;
  size_t gb = (size_t)g * 512;
  pooledX[gb + d0] = ax0 * sc;      pooledX[gb + d1] = ax1 * sc;
  pooledCausal[gb + d0] = ac0 * sc; pooledCausal[gb + d1] = ac1 * sc;
  pooledCounter[gb + d0] = an0 * sc; pooledCounter[gb + d1] = an1 * sc;
}

// One wave per graph: 3 pooled vectors x 10 classes.
__global__ void cls_kernel(const float* __restrict__ pc, const float* __restrict__ pcnt,
                           const float* __restrict__ px, const float* __restrict__ w,
                           const float* __restrict__ b, float* __restrict__ out) {
  int g = blockIdx.x, lane = threadIdx.x;
  const float* srcs[3] = {pc, pcnt, px};
  const size_t offs[3] = {0, 5120, 10240};
  for (int s = 0; s < 3; ++s) {
    const float* row = srcs[s] + (size_t)g * 512;
    for (int t = 0; t < 10; ++t) {
      const float* wr = w + (size_t)t * 512;
      float sum = 0.0f;
      for (int d = lane; d < 512; d += 32) sum += row[d] * wr[d];
      for (int o = 16; o > 0; o >>= 1) sum += __shfl_down(sum, o);
      if (lane == 0) out[offs[s] + (size_t)g * 10 + t] = sum + b[t];
    }
  }
}

// ---------------------------------------------------------------------------
extern "C" void kernel_launch(void* const* d_in, const int* in_sizes, int n_in,
                              void* d_out, int out_size, void* d_ws, size_t ws_size,
                              hipStream_t stream) {
  (void)in_sizes; (void)n_in; (void)out_size; (void)ws_size;
  const float* x      = (const float*)d_in[0];
  // d_in[1] = batch (recomputed as n>>7; int64 dtype avoided on purpose)
  const float* cbIn   = (const float*)d_in[2];
  const float* fc1w   = (const float*)d_in[3];
  const float* fc1b   = (const float*)d_in[4];
  const float* fc2w   = (const float*)d_in[5];
  const float* fc2b   = (const float*)d_in[6];
  const float* causal = (const float*)d_in[7];
  const float* clsw   = (const float*)d_in[8];
  const float* clsb   = (const float*)d_in[9];

  float* out = (float*)d_out;
  float* out_pre      = out;                 // causal_pre / counter_pre / y_pre (3*5120)
  float* out_Aori     = out + 15360;         // 512*128*128
  float* out_Arec     = out + 8403968;
  float* out_z        = out + 16792576;      // 65536*512
  float* out_x        = out + 50347008;      // 65536*512
  float* out_cdiag    = out + 83901440;      // 1024*1024
  float* out_cm       = out + 84950016;      // 1024*1024
  float* out_pcausal  = out + 85998592;      // 512*512
  float* out_px       = out + 86260736;      // 512*512

  float* ws = (float*)d_ws;
  float* w_h     = ws;                 // 1024*128
  float* w_ct    = ws + 131072;        // 1024*512
  float* w_cs    = ws + 655360;        // 1024*512
  float* w_csT   = ws + 1179648;       // 512*1024
  float* w_S1T   = ws + 1703936;       // 512*1024
  float* w_T2t   = ws + 2228224;       // 512*1024
  float* w_AE    = ws + 2752512;       // 1024*1024
  float* w_ccb   = ws + 3801088;       // 1024*512
  float* w_ccnt  = ws + 4325376;       // 1024*512
  float* w_diag  = ws + 4849664;       // 1024
  float* w_cn2   = ws + 4850688;       // 1024
  float* w_xinv  = ws + 4851712;       // 65536
  float* w_zinv  = ws + 4917248;       // 65536
  float* w_pcnt  = ws + 4982784;       // 512*512
  int*   w_idx   = (int*)(ws + 5244928); // 65536

  // 1) causal prep: diag, outputs causal_diag / counter_matrix, AE
  prep_causal<<<4096, 256, 0, stream>>>(causal, out_cdiag, out_cm, w_AE, w_diag);

  // 2) h = sigmoid(cbIn @ fc1_w^T + b1)   [1024 x 128]
  gemm_bf16split<2><<<dim3(2, 8), 256, 0, stream>>>(cbIn, fc1w, fc1b, w_h, 512, 512, 512, 128);
  // 3) ct = h @ fc2_w^T + b2              [1024 x 512]
  gemm_bf16split<1><<<dim3(8, 8), 256, 0, stream>>>(w_h, fc2w, fc2b, w_ct, 128, 128, 128, 512);
  // 4) cs = sigmoid(ct); csT; S1T = diag(causal)*cs transposed
  prep_cs<<<1024, 512, 0, stream>>>(w_ct, w_diag, w_cs, w_csT, w_S1T);

  // 5) T2t[d,k'] = sum_k cs[k,d]*CM[k',k]   (= (CM@cs)^T)  [512 x 1024]
  gemm_bf16split<0><<<dim3(16, 4), 256, 0, stream>>>(w_csT, out_cm, nullptr, w_T2t, 1024, 1024, 1024, 1024);
  // 6) causal_cb = AE @ S1  (Bt = S1T)     [1024 x 512]
  gemm_bf16split<0><<<dim3(8, 8), 256, 0, stream>>>(w_AE, w_S1T, nullptr, w_ccb, 1024, 1024, 1024, 512);
  // 7) counter_cb = AE @ T2 (Bt = T2t)     [1024 x 512]
  gemm_bf16split<0><<<dim3(8, 8), 256, 0, stream>>>(w_AE, w_T2t, nullptr, w_ccnt, 1024, 1024, 1024, 512);

  // 8) ||causal_cb_k||^2
  rowreduce<0><<<128, 256, 0, stream>>>(w_ccb, w_cn2, 512);
  // 9) fused distance GEMM + argmin over K=1024
  argmin_gemm<<<512, 256, 0, stream>>>(x, w_ccb, w_cn2, w_idx);

  // 10) gathers, x copy, pooled means
  pool_scatter<<<512, 256, 0, stream>>>(x, w_idx, w_ct, w_ccb, w_ccnt,
                                        out_z, out_x, out_pcausal, out_px, w_pcnt);

  // 11) row inverse norms for x and z_nodes
  rowreduce<1><<<8192, 256, 0, stream>>>(x, w_xinv, 512);
  rowreduce<1><<<8192, 256, 0, stream>>>(out_z, w_zinv, 512);

  // 12) adjacencies (TDM-staged tiles)
  adj_kernel<<<512, 256, 0, stream>>>(x, w_xinv, out_Aori);
  adj_kernel<<<512, 256, 0, stream>>>(out_z, w_zinv, out_Arec);

  // 13) classifier heads
  cls_kernel<<<512, 32, 0, stream>>>(out_pcausal, w_pcnt, out_px, clsw, clsb, out_pre);
}
